// WindowAttention_21474836480195
// MI455X (gfx1250) — matrix operand order repaired and checked by reading, not compiled, not run on hardware
//
#include <hip/hip_runtime.h>

typedef _Float16 f16;
typedef _Float16 v16h __attribute__((ext_vector_type(16)));
typedef _Float16 v8h  __attribute__((ext_vector_type(8)));
typedef float    v8f  __attribute__((ext_vector_type(8)));

#define IMG   56
#define CIN   192
#define CQKV  576
#define NHD   6
#define HDIM  32
#define NWIN  64      // 8x8 windows per image
#define NTOK  49      // 7x7 tokens per window
#define NPAD  64      // padded tokens per window
#define NBATCH 32
#define SCALE 0.17677669529663687f   // 32^-0.5

#define MPAD  (NBATCH * NWIN * NPAD)  // 131072 window-ordered padded rows
#define NPIX  (NBATCH * IMG * IMG)    // 100352 pixels

// ---------------------------------------------------------------------------
// CDNA5 async copy: global -> LDS, 16B per lane, per-lane addresses.
// Tracked by ASYNCcnt; in-order completion => s_wait_asynccnt 1 with two
// outstanding copies guarantees the older buffer is resident.
// ---------------------------------------------------------------------------
__device__ __forceinline__ void async_copy16(unsigned lds_addr, const void* g) {
  asm volatile("global_load_async_to_lds_b128 %0, %1, off" ::"v"(lds_addr),
               "v"((unsigned long long)(unsigned long)g)
               : "memory");
}
__device__ __forceinline__ void wait_async0() {
  asm volatile("s_wait_asynccnt 0" ::: "memory");
}
__device__ __forceinline__ void wait_async1() {
  asm volatile("s_wait_asynccnt 1" ::: "memory");
}
__device__ __forceinline__ unsigned lds_off(const void* p) {
  return (unsigned)(unsigned long)p;  // low 32 bits of generic ptr = LDS addr
}

__device__ __forceinline__ v8f wmma16(v16h a, v16h b, v8f c) {
  return __builtin_amdgcn_wmma_f32_16x16x32_f16(false, a, false, b, (short)0, c,
                                                false, false);
}

__device__ __forceinline__ v16h cat8(v8h lo, v8h hi) {
  return __builtin_shufflevector(lo, hi, 0, 1, 2, 3, 4, 5, 6, 7, 8, 9, 10, 11,
                                 12, 13, 14, 15);
}

// A fragment 16x32 f16, row-major source. lane<16: M=lane, K 0..15;
// lane>=16: M=lane-16, K 16..31.
__device__ __forceinline__ v16h load_a_frag(const f16* lds, int row0,
                                            int strideH, int lane) {
  const int r   = lane & 15;
  const int kb8 = (lane >> 4) * 16;
  const f16* base = lds + (row0 + r) * strideH;
  return cat8(*(const v8h*)(base + kb8), *(const v8h*)(base + 8 + kb8));
}

// B fragment 32x16 f16 from [N][K] row-major storage: lanes 0-15 K=0..15,
// lanes 16-31 K=16..31, N = lane&15 (per-lane K contiguous).
__device__ __forceinline__ v16h load_b_frag_nk(const f16* src, int ncol0,
                                               int strideH, int lane) {
  const int ncol = lane & 15;
  const int kb16 = (lane >> 4) * 16;
  const f16* base = src + (ncol0 + ncol) * strideH + kb16;
  return cat8(*(const v8h*)(base), *(const v8h*)(base + 8));
}

// ---------------------------------------------------------------------------
// Kernel 0: one-shot fp32 -> f16 conversion (x, w_qkv, w_out)
// ---------------------------------------------------------------------------
__global__ __launch_bounds__(256) void cvt_kernel(const float* __restrict__ in,
                                                  f16* __restrict__ out,
                                                  long n) {
  const long i = ((long)blockIdx.x * blockDim.x + threadIdx.x) * 8;
  if (i < n) {
    const float4* p = (const float4*)(in + i);
    const float4 a = p[0], b = p[1];
    v8h o;
    o[0] = (f16)a.x; o[1] = (f16)a.y; o[2] = (f16)a.z; o[3] = (f16)a.w;
    o[4] = (f16)b.x; o[5] = (f16)b.y; o[6] = (f16)b.z; o[7] = (f16)b.w;
    *(v8h*)(out + i) = o;
  }
}

// ---------------------------------------------------------------------------
// Kernel 1: fused roll + window permute + QKV GEMM
//   f16 WMMA, double-buffered async A-stage.
//   q/k/v stored [B][NH][win][64][32] f16; tok>=49 rows are zero.
// ---------------------------------------------------------------------------
__global__ __launch_bounds__(256) void qkv_kernel(
    const f16* __restrict__ xh, const f16* __restrict__ wqh,
    f16* __restrict__ qb, f16* __restrict__ kb, f16* __restrict__ vb) {
  __shared__ f16 As[2][64][40];  // double-buffered, 80B rows, 16B aligned

  const int tid  = threadIdx.x;
  const int lane = tid & 31;
  const int wave = tid >> 5;
  const int m0 = blockIdx.x * 64;  // one (batch, window) per M-block
  const int n0 = blockIdx.y * 64;
  const int tm  = wave & 3;
  const int tn0 = (wave >> 2) * 2;

  const int bb  = m0 >> 12;
  const int win = (m0 >> 6) & 63;

  // per-thread A-stage source: 8 f16 (16B) of one token row, roll fused
  const int arow = tid >> 2;         // token within window (0..63)
  const int acol = (tid & 3) * 8;
  long src = -1;
  if (arow < NTOK) {
    const int hp = (win >> 3) * 7 + arow / 7;
    const int wp = (win & 7) * 7 + arow % 7;
    const int hs = (hp + 3) % IMG;
    const int ws = (wp + 3) % IMG;
    src = ((long)(bb * IMG + hs) * IMG + ws) * CIN;
  } else {
    // zero pad rows once in both buffers; never rewritten
    *(v8h*)&As[0][arow][acol] = (v8h){};
    *(v8h*)&As[1][arow][acol] = (v8h){};
  }
  const unsigned adst0 = lds_off(&As[0][arow][acol]);
  const unsigned adst1 = lds_off(&As[1][arow][acol]);

  v8f acc0 = {}, acc1 = {};
  const int ncol = lane & 15;
  const int kb16 = (lane >> 4) * 16;
  const f16* w0p = wqh + (long)(n0 + tn0 * 16 + ncol) * CIN + kb16;
  const f16* w1p = wqh + (long)(n0 + (tn0 + 1) * 16 + ncol) * CIN + kb16;

  // software pipeline: chunk kc lives in buffer (kc/32)&1
  if (src >= 0) async_copy16(adst0, xh + src + acol);
#pragma unroll
  for (int kc = 0; kc < CIN; kc += 32) {
    const int cur = (kc >> 5) & 1;
    if (kc + 32 < CIN) {
      if (src >= 0)
        async_copy16(cur ? adst0 : adst1, xh + src + kc + 32 + acol);
      wait_async1();  // oldest (current buffer) done; next still in flight
    } else {
      wait_async0();
    }
    __syncthreads();

    const v16h a = load_a_frag(&As[cur][0][0], tm * 16, 40, lane);
    acc0 = wmma16(a, cat8(*(const v8h*)(w0p + kc), *(const v8h*)(w0p + kc + 8)),
                  acc0);
    acc1 = wmma16(a, cat8(*(const v8h*)(w1p + kc), *(const v8h*)(w1p + kc + 8)),
                  acc1);
    __syncthreads();  // protects buffer overwrite two chunks ahead
  }

  // D layout: VGPR j -> M = j + 8*(lane>=16), N = lane&15.
  // Whole block is one (b, win): rows map 1:1 to padded tokens.
  const int tok0 = tm * 16 + (lane >> 4) * 8;
#pragma unroll
  for (int t = 0; t < 2; ++t) {
    const v8f acc = t ? acc1 : acc0;
    const int gc = n0 + (tn0 + t) * 16 + ncol;
    const int kind = gc / CIN;
    const int head = (gc % CIN) / HDIM;
    const int hd = gc & (HDIM - 1);
    f16* dst = (kind == 0) ? qb : ((kind == 1) ? kb : vb);
    const long row0 = ((long)(bb * NHD + head) * NWIN + win) * NPAD + tok0;
#pragma unroll
    for (int j = 0; j < 8; ++j) dst[(row0 + j) * HDIM + hd] = (f16)acc[j];
  }
}

// ---------------------------------------------------------------------------
// Kernel 2: windowed attention. One block per (b, head, window), 4 waves.
// ---------------------------------------------------------------------------
__global__ __launch_bounds__(128) void attn_kernel(
    const f16* __restrict__ qb, const f16* __restrict__ kb,
    const f16* __restrict__ vb, const float* __restrict__ pos,
    f16* __restrict__ ob) {
  __shared__ f16 Qs[NPAD * HDIM];   // [tok][d]
  __shared__ f16 Ks[NPAD * HDIM];   // [tok][d]
  __shared__ f16 VT[HDIM * NPAD];   // [d][tok]  (transposed at load)
  __shared__ float Ss[NPAD * NPAD]; // scores fp32
  __shared__ f16 Ps[NPAD * NPAD];   // softmax probs f16
  __shared__ float Pos[169];        // 13x13 relative bias table

  const int tid  = threadIdx.x;
  const int lane = tid & 31;
  const int wave = tid >> 5;
  const int bid  = blockIdx.x;                  // = (b*6 + head)*64 + win
  const int win  = bid & 63;
  const int head = (bid >> 6) % NHD;
  const int b    = bid / (NHD * NWIN);
  const long base = (long)bid * (NPAD * HDIM);

  // stage q, k via async-to-LDS (4KB each); v transposed through VGPRs
  {
    const char* qg = (const char*)(qb + base);
    const char* kg = (const char*)(kb + base);
    const unsigned qd = lds_off(Qs), kd = lds_off(Ks);
    async_copy16(qd + tid * 16, qg + tid * 16);
    async_copy16(qd + 2048 + tid * 16, qg + 2048 + tid * 16);
    async_copy16(kd + tid * 16, kg + tid * 16);
    async_copy16(kd + 2048 + tid * 16, kg + 2048 + tid * 16);

    const int row = tid >> 1;
    const int c0 = (tid & 1) * 16;
    const f16* vg = vb + base + row * HDIM + c0;
    const v8h v0 = *(const v8h*)(vg);
    const v8h v1 = *(const v8h*)(vg + 8);
#pragma unroll
    for (int e = 0; e < 8; ++e) {
      VT[(c0 + e) * NPAD + row] = v0[e];
      VT[(c0 + 8 + e) * NPAD + row] = v1[e];
    }
    for (int e = tid; e < 169; e += 128) Pos[e] = pos[e];
    wait_async0();
  }
  __syncthreads();

  // S = q @ k^T : wave owns row-tile `wave`; K=32 in a single WMMA
  {
    const v16h aq = load_a_frag(Qs, wave * 16, HDIM, lane);
    const int ncol = lane & 15;
#pragma unroll
    for (int tn = 0; tn < 4; ++tn) {
      const v16h bk = load_b_frag_nk(Ks, tn * 16, HDIM, lane);
      v8f z = {};
      const v8f s = wmma16(aq, bk, z);
#pragma unroll
      for (int j = 0; j < 8; ++j)
        Ss[(wave * 16 + j + (lane >> 4) * 8) * NPAD + tn * 16 + ncol] = s[j];
    }
  }
  __syncthreads();

  // scale + relative position bias + shift masks; pad cols -> -inf
  {
    const bool ul = ((win >> 3) == 7);
    const bool lr = ((win & 7) == 7);
    for (int e = tid; e < NPAD * NPAD; e += 128) {
      const int i = e >> 6, j = e & 63;
      float val;
      if (i < NTOK && j < NTOK) {
        val = Ss[e] * SCALE;
        const int xi = i / 7, yi = i % 7, xj = j / 7, yj = j % 7;
        val += Pos[(xj - xi + 6) * 13 + (yj - yi + 6)];
        if (ul && ((i >= 28) != (j >= 28))) val = -1e30f;
        if (lr && ((yi >= 4) != (yj >= 4))) val = -1e30f;
      } else {
        val = (j < NTOK) ? 0.0f : -1e30f;
      }
      Ss[e] = val;
    }
  }
  __syncthreads();

  // row softmax, emit f16 probabilities
  if (tid < NPAD) {
    float* row = &Ss[tid * NPAD];
    float m = -1e30f;
#pragma unroll 8
    for (int j = 0; j < NPAD; ++j) m = fmaxf(m, row[j]);
    float sum = 0.f;
#pragma unroll 8
    for (int j = 0; j < NPAD; ++j) {
      const float ex = __expf(row[j] - m);
      sum += ex;
      row[j] = ex;
    }
    const float inv = 1.0f / sum;
    f16* prow = &Ps[tid * NPAD];
#pragma unroll 8
    for (int j = 0; j < NPAD; ++j) prow[j] = (f16)(row[j] * inv);
  }
  __syncthreads();

  // O = P @ v : K=64 chained, 2 N-tiles of HD
  {
    const long orow0 = ((long)b * NWIN + win) * NPAD;
    const int ncol = lane & 15;
#pragma unroll
    for (int tnc = 0; tnc < 2; ++tnc) {
      v8f o = {};
#pragma unroll
      for (int kk = 0; kk < 2; ++kk) {
        const v16h ap = load_a_frag(Ps + kk * 32, wave * 16, NPAD, lane);
        const v16h bv = load_b_frag_nk(VT + kk * 32, tnc * 16, NPAD, lane);
        o = wmma16(ap, bv, o);
      }
#pragma unroll
      for (int j = 0; j < 8; ++j) {
        const int i = wave * 16 + j + (lane >> 4) * 8;
        ob[(orow0 + i) * CIN + head * HDIM + tnc * 16 + ncol] = (f16)o[j];
      }
    }
  }
}

// ---------------------------------------------------------------------------
// Kernel 3: output projection + bias, fused inverse window permute + roll(+3)
//   double-buffered async A-stage.
// ---------------------------------------------------------------------------
__global__ __launch_bounds__(256) void proj_kernel(
    const f16* __restrict__ ao, const f16* __restrict__ woh,
    const float* __restrict__ bout, float* __restrict__ out) {
  __shared__ f16 As[2][64][40];

  const int tid  = threadIdx.x;
  const int lane = tid & 31;
  const int wave = tid >> 5;
  const int m0 = blockIdx.x * 64;
  const int n0 = blockIdx.y * 64;
  const int tm  = wave & 3;
  const int tn0 = (wave >> 2) * 2;

  // A gather: output pixel (b,h,w) <- window row of pre-roll pixel
  const int arow = tid >> 2;
  const int acol = (tid & 3) * 8;
  const int gr = m0 + arow;
  const int b = gr / (IMG * IMG);
  const int p = gr % (IMG * IMG);
  const int h = p / IMG, w = p % IMG;
  const int hs = (h + IMG - 3) % IMG;
  const int ws = (w + IMG - 3) % IMG;
  const f16* asrc = ao + ((long)b * (NWIN * NPAD) +
                          ((hs / 7) * 8 + ws / 7) * NPAD + (hs % 7) * 7 +
                          (ws % 7)) * CIN + acol;
  const unsigned adst0 = lds_off(&As[0][arow][acol]);
  const unsigned adst1 = lds_off(&As[1][arow][acol]);

  v8f acc0 = {}, acc1 = {};
  const int ncol = lane & 15;
  const int kb16 = (lane >> 4) * 16;
  const f16* w0p = woh + (long)(n0 + tn0 * 16 + ncol) * CIN + kb16;
  const f16* w1p = woh + (long)(n0 + (tn0 + 1) * 16 + ncol) * CIN + kb16;

  async_copy16(adst0, asrc);
#pragma unroll
  for (int kc = 0; kc < CIN; kc += 32) {
    const int cur = (kc >> 5) & 1;
    if (kc + 32 < CIN) {
      async_copy16(cur ? adst0 : adst1, asrc + kc + 32);
      wait_async1();
    } else {
      wait_async0();
    }
    __syncthreads();

    const v16h a = load_a_frag(&As[cur][0][0], tm * 16, 40, lane);
    acc0 = wmma16(a, cat8(*(const v8h*)(w0p + kc), *(const v8h*)(w0p + kc + 8)),
                  acc0);
    acc1 = wmma16(a, cat8(*(const v8h*)(w1p + kc), *(const v8h*)(w1p + kc + 8)),
                  acc1);
    __syncthreads();
  }

  const int drow0 = m0 + tm * 16 + (lane >> 4) * 8;
#pragma unroll
  for (int t = 0; t < 2; ++t) {
    const v8f acc = t ? acc1 : acc0;
    const int gc = n0 + (tn0 + t) * 16 + ncol;
    const float bias = bout[gc];
#pragma unroll
    for (int j = 0; j < 8; ++j)
      out[(long)(drow0 + j) * CIN + gc] = acc[j] + bias;
  }
}

extern "C" void kernel_launch(void* const* d_in, const int* in_sizes, int n_in,
                              void* d_out, int out_size, void* d_ws,
                              size_t ws_size, hipStream_t stream) {
  const float* x    = (const float*)d_in[0];
  const float* wqkv = (const float*)d_in[1];
  const float* pos  = (const float*)d_in[2];
  const float* wout = (const float*)d_in[3];
  const float* bout = (const float*)d_in[4];
  float* out = (float*)d_out;

  const size_t xelems = (size_t)NPIX * CIN;                        // 19.3M
  const size_t qelems = (size_t)NBATCH * NHD * NWIN * NPAD * HDIM; // 25.2M
  f16* ws   = (f16*)d_ws;
  f16* xh   = ws;
  f16* wqh  = xh + xelems;
  f16* woh  = wqh + (size_t)CQKV * CIN;
  f16* qb   = woh + (size_t)CIN * CIN;
  f16* kb   = qb + qelems;
  f16* vb   = kb + qelems;
  f16* ao   = vb + qelems;  // [MPAD][192] f16

  // one-shot precision conversion (amortized: ~120MB extra traffic total)
  cvt_kernel<<<dim3((unsigned)(xelems / 2048)), 256, 0, stream>>>(x, xh,
                                                                  (long)xelems);
  cvt_kernel<<<dim3((CQKV * CIN) / 2048), 256, 0, stream>>>(
      wqkv, wqh, (long)(CQKV * CIN));
  cvt_kernel<<<dim3((CIN * CIN) / 2048), 256, 0, stream>>>(wout, woh,
                                                           (long)(CIN * CIN));

  qkv_kernel<<<dim3(MPAD / 64, CQKV / 64), 256, 0, stream>>>(xh, wqh, qb, kb,
                                                             vb);
  attn_kernel<<<dim3(NBATCH * NHD * NWIN), 128, 0, stream>>>(qb, kb, vb, pos,
                                                             ao);
  proj_kernel<<<dim3(NPIX / 64, CIN / 64), 256, 0, stream>>>(ao, woh, bout,
                                                             out);
}
